// MultiHeadLatentAttention_52432960749837
// MI455X (gfx1250) — matrix operand order repaired and checked by reading, not compiled, not run on hardware
//
#include <hip/hip_runtime.h>

#define BB   2
#define TT   2048
#define DD   2048
#define NH   16
#define HD   128
#define QL   1024
#define KVL  512
#define MTOK (BB * TT)

typedef __attribute__((ext_vector_type(16))) __bf16 v16bf;
typedef __attribute__((ext_vector_type(8)))  float  v8f;
typedef __attribute__((ext_vector_type(4)))  unsigned int v4u;
typedef __attribute__((ext_vector_type(8)))  int    v8i;
typedef __attribute__((ext_vector_type(4)))  int    v4i;

union BF16x16 { v16bf v; unsigned short s[16]; };

__device__ __forceinline__ unsigned short f2bf(float f) {
    unsigned int u = __float_as_uint(f);
    u += 0x7FFFu + ((u >> 16) & 1u);   // round-to-nearest-even
    return (unsigned short)(u >> 16);
}
__device__ __forceinline__ float bf2f(unsigned short b) {
    return __uint_as_float((unsigned int)b << 16);
}

// 16-lane (half-wave) reductions via ds_swizzle_b32 XOR patterns (wave32-safe;
// masks 1,2,4,8 never cross the lane-16 boundary of a C-fragment row group).
template <int IMM>
__device__ __forceinline__ float swz(float x) {
    return __int_as_float(__builtin_amdgcn_ds_swizzle(__float_as_int(x), IMM));
}
__device__ __forceinline__ float rowmax16(float x) {
    x = fmaxf(x, swz<0x041F>(x));
    x = fmaxf(x, swz<0x081F>(x));
    x = fmaxf(x, swz<0x101F>(x));
    x = fmaxf(x, swz<0x201F>(x));
    return x;
}
__device__ __forceinline__ float rowsum16(float x) {
    x += swz<0x041F>(x);
    x += swz<0x081F>(x);
    x += swz<0x101F>(x);
    x += swz<0x201F>(x);
    return x;
}

// ---------------------------------------------------------------------------
// TDM: issue a 2D tile load (tile_w x tile_h bf16, row-major, row stride =
// dim0 elements) from global to LDS. D# packed per ISA 8.3/8.4:
//   group0: count=1 | lds_addr | global_addr[56:0] | type=2
//   group1: data_size=1 (2B), tensor_dim0=dim0, tensor_dim1=rows,
//           tile_dim0=tile_w, tile_dim1=tile_h, dim0_stride=dim0
// 6-arg builtin form (clang-23 / therock-10.0 headers).
// ---------------------------------------------------------------------------
__device__ __forceinline__ void tdm_load_2d(
    unsigned lds_off, const unsigned short* gptr,
    int dim0, int rows_total, int tile_w, int tile_h)
{
    unsigned long long ga = (unsigned long long)(uintptr_t)gptr;
    v4u g0 = { 1u,                                  // count=1, no gather
               lds_off,
               (unsigned)ga,
               (unsigned)((ga >> 32) & 0x1FFFFFFull) | (2u << 30) }; // type=2
    v8i g1 = { (int)(1u << 16),                                    // data_size=2B
               (int)(((unsigned)dim0 & 0xFFFFu) << 16),            // dim0 lo16
               (int)(((unsigned)dim0 >> 16) |
                     (((unsigned)rows_total & 0xFFFFu) << 16)),    // dim0 hi / dim1 lo
               (int)((((unsigned)rows_total >> 16) & 0xFFFFu) |
                     ((unsigned)tile_w << 16)),                    // dim1 hi / tile_dim0
               (int)((unsigned)tile_h & 0xFFFFu),                  // tile_dim1 (tile_dim2=0)
               dim0,                                               // dim0_stride lo32
               0, 0 };
    v4i z4 = { 0, 0, 0, 0 };
    v8i z8 = { 0, 0, 0, 0, 0, 0, 0, 0 };
    __builtin_amdgcn_tensor_load_to_lds(g0, g1, z4, z4, z8, 0);
}

// ---------------------------------------------------------------------------
// Bulk f32 -> bf16 conversion, 8 elements/thread, vectorized 128b loads/stores.
// n must be a multiple of 2048 (true for every buffer here).
// ---------------------------------------------------------------------------
__global__ __launch_bounds__(256) void cvt_f32_bf16(
    const float* __restrict__ src, unsigned short* __restrict__ dst)
{
    size_t i = ((size_t)blockIdx.x * 256 + threadIdx.x) * 8;
    float4 a = *(const float4*)(src + i);
    float4 b = *(const float4*)(src + i + 4);
    uint4 o;
    o.x = (unsigned)f2bf(a.x) | ((unsigned)f2bf(a.y) << 16);
    o.y = (unsigned)f2bf(a.z) | ((unsigned)f2bf(a.w) << 16);
    o.z = (unsigned)f2bf(b.x) | ((unsigned)f2bf(b.y) << 16);
    o.w = (unsigned)f2bf(b.z) | ((unsigned)f2bf(b.w) << 16);
    *(uint4*)(dst + i) = o;
}

// ---------------------------------------------------------------------------
// TN GEMM on bf16: C[M,N] = A[M,K] * W[N,K]^T, f32 accumulation via
// v_wmma_f32_16x16x32_bf16. Block tile 128x128, 8 waves (2x4), per-wave
// 64x32 (4x2 WMMA tiles). Tiles staged to LDS by the Tensor Data Mover,
// double-buffered: wave 0 issues tile k+1 while all waves compute tile k.
// Output f32 (final projection) or bf16 (chained activations).
// ---------------------------------------------------------------------------
template <bool F32OUT>
__global__ __launch_bounds__(256) void gemm_bf16_wmma(
    const unsigned short* __restrict__ A, const unsigned short* __restrict__ W,
    void* __restrict__ Cp, int M, int N, int K)
{
    __shared__ __align__(16) unsigned short lds_a[2][128 * 32];
    __shared__ __align__(16) unsigned short lds_b[2][128 * 32];

    const int tid  = threadIdx.x;
    const int lane = tid & 31;
    const int wave = tid >> 5;
    const int l15  = lane & 15;
    const int kh   = lane >> 4;
    const int wm   = wave >> 2;      // 0..1
    const int wn   = wave & 3;       // 0..3
    const int mBase = blockIdx.y * 128;
    const int nBase = blockIdx.x * 128;

    v8f acc[4][2];
#pragma unroll
    for (int i = 0; i < 4; ++i)
#pragma unroll
        for (int j = 0; j < 2; ++j)
            acc[i][j] = (v8f){0.f, 0.f, 0.f, 0.f, 0.f, 0.f, 0.f, 0.f};

    const int nk = K / 32;
    if (wave == 0) {
        tdm_load_2d((unsigned)(uintptr_t)&lds_a[0][0], A + (size_t)mBase * K,
                    K, M, 32, 128);
        tdm_load_2d((unsigned)(uintptr_t)&lds_b[0][0], W + (size_t)nBase * K,
                    K, N, 32, 128);
    }

    for (int kt = 0; kt < nk; ++kt) {
        const int cur = kt & 1;
        if (wave == 0) {
            if (kt + 1 < nk) {
                const int k1 = (kt + 1) * 32;
                tdm_load_2d((unsigned)(uintptr_t)&lds_a[cur ^ 1][0],
                            A + (size_t)mBase * K + k1, K, M, 32, 128);
                tdm_load_2d((unsigned)(uintptr_t)&lds_b[cur ^ 1][0],
                            W + (size_t)nBase * K + k1, K, N, 32, 128);
                __builtin_amdgcn_s_wait_tensorcnt(2);  // oldest pair done
            } else {
                __builtin_amdgcn_s_wait_tensorcnt(0);
            }
        }
        __syncthreads();                               // tile `cur` ready

        BF16x16 af[4];
#pragma unroll
        for (int ms = 0; ms < 4; ++ms) {
            int ml = wm * 64 + ms * 16 + l15;
#pragma unroll
            for (int e = 0; e < 16; ++e) {
                int k = kh * 16 + e;
                af[ms].s[e] = lds_a[cur][ml * 32 + k];
            }
        }
        BF16x16 bf[2];
#pragma unroll
        for (int ns = 0; ns < 2; ++ns) {
            int nl = wn * 32 + ns * 16 + l15;
#pragma unroll
            for (int e = 0; e < 16; ++e)
                bf[ns].s[e] = lds_b[cur][nl * 32 + kh * 16 + e];
        }
#pragma unroll
        for (int ms = 0; ms < 4; ++ms)
#pragma unroll
            for (int ns = 0; ns < 2; ++ns)
                acc[ms][ns] = __builtin_amdgcn_wmma_f32_16x16x32_bf16(
                    false, af[ms].v, false, bf[ns].v,
                    (short)0, acc[ms][ns], false, false);

        __syncthreads();                               // reads done before overwrite
    }

#pragma unroll
    for (int ms = 0; ms < 4; ++ms)
#pragma unroll
        for (int ns = 0; ns < 2; ++ns) {
            int n = nBase + wn * 32 + ns * 16 + l15;
#pragma unroll
            for (int r = 0; r < 8; ++r) {
                int m = mBase + wm * 64 + ms * 16 + r + 8 * kh;
                if (F32OUT)
                    ((float*)Cp)[(size_t)m * N + n] = acc[ms][ns][r];
                else
                    ((unsigned short*)Cp)[(size_t)m * N + n] = f2bf(acc[ms][ns][r]);
            }
        }
}

// ---------------------------------------------------------------------------
// Fused per-head RMSNorm + RoPE on bf16 activations, in place.
// grid = MTOK*NH blocks of 128; buf layout (B*T, NH*HD) token-major.
// ---------------------------------------------------------------------------
__global__ __launch_bounds__(128) void rmsnorm_rope(
    unsigned short* __restrict__ buf, const float* __restrict__ w)
{
    const int idx   = blockIdx.x;
    const int token = idx / NH;
    const int head  = idx % NH;
    const int t     = token % TT;
    unsigned short* p = buf + (size_t)token * DD + head * HD;
    const int i = threadIdx.x;

    float v  = bf2f(p[i]);
    float vp = bf2f(p[i ^ 64]);        // rotate_half partner (read phase)

    __shared__ float red[128];
    red[i] = v * v;
    __syncthreads();
#pragma unroll
    for (int st = 64; st > 0; st >>= 1) {
        if (i < st) red[i] += red[i + st];
        __syncthreads();
    }
    float rsq = rsqrtf(red[0] * (1.f / (float)HD) + 1e-6f);

    float xn  = v  * rsq * w[i];
    float xnp = vp * rsq * w[i ^ 64];

    int   j        = i & 63;
    float inv_freq = __expf(-((float)(2 * j) * (1.f / (float)HD)) * 9.210340371976184f);
    float ang = (float)t * inv_freq;
    float sv = sinf(ang), cv = cosf(ang);
    float rot = (i < 64) ? -xnp : xnp;
    float outv = xn * cv + rot * sv;

    __syncthreads();                   // partner reads complete before writes
    p[i] = f2bf(outv);
}

// ---------------------------------------------------------------------------
// Flash attention (causal), bf16 Q/K/V in, bf16 out, f32 accumulation.
// grid = (T/128, B*NH), 8 waves/block; each wave owns 16 q rows.
// ---------------------------------------------------------------------------
__global__ __launch_bounds__(256) void flash_attn_wmma(
    const unsigned short* __restrict__ Q, const unsigned short* __restrict__ Kb,
    const unsigned short* __restrict__ V, unsigned short* __restrict__ O)
{
    __shared__ __align__(16) unsigned short lds_k[32 * 128];
    __shared__ __align__(16) unsigned short lds_v[32 * 128];
    __shared__ __align__(16) unsigned short lds_p[8 * 16 * 32];

    const int tid  = threadIdx.x;
    const int lane = tid & 31;
    const int wave = tid >> 5;
    const int l15  = lane & 15;
    const int kh   = lane >> 4;
    const int b    = blockIdx.y / NH;
    const int h    = blockIdx.y % NH;
    const int qbase = blockIdx.x * 128;
    const int wrow0 = qbase + wave * 16;
    const size_t tok0 = (size_t)b * TT;
    const float scl = 0.08838834764831845f; // 1/sqrt(128)

    // Q fragments: per chunk, two contiguous 8-element runs -> 2x b128 loads.
    BF16x16 qf[4];
    {
        const unsigned short* qp = Q + (tok0 + wrow0 + l15) * (size_t)DD + h * HD;
        const uint4* qv = (const uint4*)qp;
#pragma unroll
        for (int kc = 0; kc < 4; ++kc) {
            int k0 = kc * 32 + (kh ? 8 : 0);
            *(uint4*)&qf[kc].s[0] = qv[k0 >> 3];
            *(uint4*)&qf[kc].s[8] = qv[(k0 + 16) >> 3];
        }
    }

    v8f o[8];
    float rm[8], rs[8];
#pragma unroll
    for (int c = 0; c < 8; ++c) o[c] = (v8f){0.f,0.f,0.f,0.f,0.f,0.f,0.f,0.f};
#pragma unroll
    for (int r = 0; r < 8; ++r) { rm[r] = -INFINITY; rs[r] = 0.f; }

    const int nblk = blockIdx.x * 4 + 4;
    for (int jb = 0; jb < nblk; ++jb) {
        const int j0 = jb * 32;
        __syncthreads();
        // Stage 32 keys/values (bf16) with 128-bit copies: 512 uint4 each.
#pragma unroll
        for (int t = 0; t < 2; ++t) {
            int u = tid + t * 256;
            int r = u >> 4, c = (u & 15) * 8;
            size_t g = (tok0 + j0 + r) * (size_t)DD + h * HD + c;
            ((uint4*)lds_k)[u] = *(const uint4*)(Kb + g);
            ((uint4*)lds_v)[u] = *(const uint4*)(V + g);
        }
        __syncthreads();
        if (j0 > wrow0 + 15) continue;          // tile fully masked for this wave

        // S(16x32) = Q(16x128) * K^T
        v8f s[2];
        s[0] = (v8f){0.f,0.f,0.f,0.f,0.f,0.f,0.f,0.f};
        s[1] = (v8f){0.f,0.f,0.f,0.f,0.f,0.f,0.f,0.f};
#pragma unroll
        for (int ns = 0; ns < 2; ++ns) {
            int key = ns * 16 + l15;
#pragma unroll
            for (int kc = 0; kc < 4; ++kc) {
                BF16x16 bfr;
#pragma unroll
                for (int e = 0; e < 16; ++e)
                    bfr.s[e] = lds_k[key * 128 + kc * 32 + ((e < 8) ? 0 : 16) + kh * 8 + (e & 7)];
                s[ns] = __builtin_amdgcn_wmma_f32_16x16x32_bf16(
                    false, qf[kc].v, false, bfr.v, (short)0, s[ns], false, false);
            }
        }

        // Online softmax (row m = r + 8*kh shared by 16 lanes).
#pragma unroll
        for (int r = 0; r < 8; ++r) {
            int qq = wrow0 + r + 8 * kh;
            float v0 = s[0][r] * scl; if (j0 + l15      > qq) v0 = -INFINITY;
            float v1 = s[1][r] * scl; if (j0 + 16 + l15 > qq) v1 = -INFINITY;
            float mx    = rowmax16(fmaxf(v0, v1));
            float mnew  = fmaxf(rm[r], mx);
            float alpha = __expf(rm[r] - mnew);
            rm[r] = mnew;
            float p0 = __expf(v0 - mnew);
            float p1 = __expf(v1 - mnew);
            rs[r] = rs[r] * alpha + rowsum16(p0 + p1);
#pragma unroll
            for (int c = 0; c < 8; ++c) o[c][r] *= alpha;
            int m = r + 8 * kh;
            lds_p[wave * 512 + m * 32 +      l15] = f2bf(p0);
            lds_p[wave * 512 + m * 32 + 16 + l15] = f2bf(p1);
        }

        // P back as A-fragment (layout swap through per-wave LDS scratch).
        BF16x16 pf;
#pragma unroll
        for (int e = 0; e < 16; ++e) {
            int k = kh * 16 + e;
            pf.s[e] = lds_p[wave * 512 + l15 * 32 + k];
        }

        // O(16x128) += P(16x32) * V(32x128)
#pragma unroll
        for (int c = 0; c < 8; ++c) {
            BF16x16 vf;
#pragma unroll
            for (int e = 0; e < 16; ++e)
                vf.s[e] = lds_v[(kh * 16 + e) * 128 + c * 16 + l15];
            o[c] = __builtin_amdgcn_wmma_f32_16x16x32_bf16(
                false, pf.v, false, vf.v, (short)0, o[c], false, false);
        }
    }

#pragma unroll
    for (int r = 0; r < 8; ++r) {
        float inv = 1.f / rs[r];
        size_t row = (tok0 + wrow0 + r + 8 * kh) * (size_t)DD + h * HD;
#pragma unroll
        for (int c = 0; c < 8; ++c)
            O[row + c * 16 + l15] = f2bf(o[c][r] * inv);
    }
}

// ---------------------------------------------------------------------------
extern "C" void kernel_launch(void* const* d_in, const int* in_sizes, int n_in,
                              void* d_out, int out_size, void* d_ws, size_t ws_size,
                              hipStream_t stream) {
    (void)in_sizes; (void)n_in; (void)out_size; (void)ws_size;
    const float* x      = (const float*)d_in[0];
    /* d_in[1] = attn_mask: causal, applied analytically */
    const float* q_a_w  = (const float*)d_in[2];
    const float* q_b_w  = (const float*)d_in[3];
    const float* kv_a_w = (const float*)d_in[4];
    const float* k_b_w  = (const float*)d_in[5];
    const float* v_b_w  = (const float*)d_in[6];
    const float* o_w    = (const float*)d_in[7];
    const float* q_n_w  = (const float*)d_in[8];
    const float* k_n_w  = (const float*)d_in[9];
    float* out = (float*)d_out;

    unsigned short* u = (unsigned short*)d_ws;
    unsigned short* xb    = u; u += (size_t)MTOK * DD;
    unsigned short* wqa   = u; u += (size_t)QL   * DD;
    unsigned short* wqb   = u; u += (size_t)DD   * QL;
    unsigned short* wkva  = u; u += (size_t)KVL  * DD;
    unsigned short* wkb   = u; u += (size_t)DD   * KVL;
    unsigned short* wvb   = u; u += (size_t)DD   * KVL;
    unsigned short* wo    = u; u += (size_t)DD   * DD;
    unsigned short* qlat  = u; u += (size_t)MTOK * QL;
    unsigned short* qb    = u; u += (size_t)MTOK * DD;
    unsigned short* kvb   = u; u += (size_t)MTOK * KVL;
    unsigned short* kb    = u; u += (size_t)MTOK * DD;
    unsigned short* vb    = u; u += (size_t)MTOK * DD;
    unsigned short* attnb = u;

    dim3 blk(256);
    // one-time f32 -> bf16 conversions (2048 elements per block)
    cvt_f32_bf16<<<dim3((MTOK * DD) / 2048), blk, 0, stream>>>(x,      xb);
    cvt_f32_bf16<<<dim3((QL   * DD) / 2048), blk, 0, stream>>>(q_a_w,  wqa);
    cvt_f32_bf16<<<dim3((DD   * QL) / 2048), blk, 0, stream>>>(q_b_w,  wqb);
    cvt_f32_bf16<<<dim3((KVL  * DD) / 2048), blk, 0, stream>>>(kv_a_w, wkva);
    cvt_f32_bf16<<<dim3((DD  * KVL) / 2048), blk, 0, stream>>>(k_b_w,  wkb);
    cvt_f32_bf16<<<dim3((DD  * KVL) / 2048), blk, 0, stream>>>(v_b_w,  wvb);
    cvt_f32_bf16<<<dim3((DD   * DD) / 2048), blk, 0, stream>>>(o_w,    wo);

    gemm_bf16_wmma<false><<<dim3(QL  / 128, MTOK / 128), blk, 0, stream>>>(xb,   wqa,  qlat,  MTOK, QL,  DD);
    gemm_bf16_wmma<false><<<dim3(DD  / 128, MTOK / 128), blk, 0, stream>>>(qlat, wqb,  qb,    MTOK, DD,  QL);
    gemm_bf16_wmma<false><<<dim3(KVL / 128, MTOK / 128), blk, 0, stream>>>(xb,   wkva, kvb,   MTOK, KVL, DD);
    gemm_bf16_wmma<false><<<dim3(DD  / 128, MTOK / 128), blk, 0, stream>>>(kvb,  wkb,  kb,    MTOK, DD,  KVL);
    gemm_bf16_wmma<false><<<dim3(DD  / 128, MTOK / 128), blk, 0, stream>>>(kvb,  wvb,  vb,    MTOK, DD,  KVL);
    rmsnorm_rope<<<dim3(MTOK * NH), dim3(128), 0, stream>>>(qb, q_n_w);
    rmsnorm_rope<<<dim3(MTOK * NH), dim3(128), 0, stream>>>(kb, k_n_w);
    flash_attn_wmma<<<dim3(TT / 128, BB * NH), blk, 0, stream>>>(qb, kb, vb, attnb);
    gemm_bf16_wmma<true><<<dim3(DD / 128, MTOK / 128), blk, 0, stream>>>(attnb, wo, out, MTOK, DD, DD);
}
